// GQA_68539088110251
// MI455X (gfx1250) — matrix operand order repaired and checked by reading, not compiled, not run on hardware
//
#include <hip/hip_runtime.h>

// ---------------------------------------------------------------------------
// GQA attention for gfx1250 (CDNA5), bf16 WMMA everywhere, flash-style softmax
// with async (ASYNCcnt) double-buffered K/V staging into LDS.
// B=2, P=2048, C=768, H=12, KV=4, d=64, G=3.
// ---------------------------------------------------------------------------

typedef __attribute__((ext_vector_type(16))) __bf16        v16bf;
typedef __attribute__((ext_vector_type(8)))  float         v8f;
typedef __attribute__((ext_vector_type(8)))  unsigned int  v8u;

#define B_SZ   2
#define P_SEQ  2048
#define CDIM   768
#define NHEAD  12
#define NKV    4
#define GRP    3
#define HD     64
#define KVDIM  256   // NKV*HD
#define MTOT   4096  // B*P

// float -> bf16, round-to-nearest-even
__device__ __forceinline__ unsigned short f2bf(float f) {
    unsigned int u = __builtin_bit_cast(unsigned int, f);
    unsigned int r = u + 0x7FFFu + ((u >> 16) & 1u);
    return (unsigned short)(r >> 16);
}

__device__ __forceinline__ v8f vzero8() {
    v8f z = {0.f, 0.f, 0.f, 0.f, 0.f, 0.f, 0.f, 0.f};
    return z;
}

// A-matrix fragment (16x32 bf16, MxK): lane(0..15)=row M, lanes 16..31 same
// rows; per-lane K chunk: VGPR0-7 <- K [lh*16, lh*16+16) (contiguous)
__device__ __forceinline__ v16bf load_a_frag(const unsigned short* rowbase, int lh) {
    uint4 lo = *reinterpret_cast<const uint4*>(rowbase + lh * 16);
    uint4 hi = *reinterpret_cast<const uint4*>(rowbase + lh * 16 + 8);
    v8u u = {lo.x, lo.y, lo.z, lo.w, hi.x, hi.y, hi.z, hi.w};
    return __builtin_bit_cast(v16bf, u);
}

// B-matrix fragment (32x16 bf16, KxN): lane&15 = column N; lanes 0-15 hold
// K 0..15, lanes 16-31 hold K 16..31 (contiguous 32B per lane).
// Caller passes (column base in B^T row-major) + lh*16.
__device__ __forceinline__ v16bf load_b_frag(const unsigned short* p) {
    v8u u = *reinterpret_cast<const v8u*>(p);
    return __builtin_bit_cast(v16bf, u);
}

__device__ __forceinline__ v8f wmma_bf16(v16bf a, v16bf b, v8f c) {
    return __builtin_amdgcn_wmma_f32_16x16x32_bf16(
        false, a, false, b, (short)0, c, false, false);
}

// ---------------------------------------------------------------------------
// Async copy of one 32-key K/V chunk into LDS (fire-and-forget, ASYNCcnt).
//   K chunk: 32 keys x 64 halves (d)   = 4 KB  -> sk
//   V chunk: 64 rows (d) x 32 keys     = 4 KB  -> sv
// Each global_load_async_to_lds_b128 moves 16B per lane (512B per wave-issue),
// so 8 issues per matrix. LDS byte addresses = low 32 bits of generic pointer.
// ---------------------------------------------------------------------------
__device__ __forceinline__ void async_copy_kv_chunk(
    const unsigned short* __restrict__ kbase,   // + key*KVDIM is one key row
    const unsigned short* __restrict__ vbase,   // + j*P_SEQ is one d-row
    unsigned short* sk, unsigned short* sv, int kb0, int lane)
{
    const int k_row = lane >> 3, k_col = lane & 7;   // K: 4 keys / issue
    const int v_row = lane >> 2, v_col = lane & 3;   // V: 8 rows / issue
    unsigned skl = (unsigned)(size_t)sk + (unsigned)lane * 16u;
    unsigned svl = (unsigned)(size_t)sv + (unsigned)lane * 16u;
#pragma unroll
    for (int i = 0; i < 8; ++i) {
        const unsigned short* g =
            kbase + (size_t)(kb0 + i * 4 + k_row) * KVDIM + k_col * 8;
        asm volatile("global_load_async_to_lds_b128 %0, %1, off"
                     :: "v"(skl + (unsigned)i * 512u), "v"(g) : "memory");
    }
#pragma unroll
    for (int i = 0; i < 8; ++i) {
        const unsigned short* g =
            vbase + (size_t)(i * 8 + v_row) * P_SEQ + kb0 + v_col * 8;
        asm volatile("global_load_async_to_lds_b128 %0, %1, off"
                     :: "v"(svl + (unsigned)i * 512u), "v"(g) : "memory");
    }
}

// ---------------------------------------------------------------------------
// Prep kernels
// ---------------------------------------------------------------------------
__global__ void conv_bf16_kernel(const float* __restrict__ X,
                                 unsigned short* __restrict__ Xb, int n) {
    int i = blockIdx.x * blockDim.x + threadIdx.x;
    if (i < n) Xb[i] = f2bf(X[i]);
}

// W [Kin,Nout] f32 row-major  ->  Wt [Nout,Kin] bf16 row-major,
// with optional head-permutation of the input (row) axis:
// Wt[n, perm[h]*64+j] = W[h*64+j, n]   (folds xs = gather(x, perm) into W).
__global__ void prep_w_kernel(const float* __restrict__ W,
                              unsigned short* __restrict__ Wt,
                              const int* __restrict__ perm,
                              int Kin, int Nout, int use_perm) {
    int idx = blockIdx.x * blockDim.x + threadIdx.x;
    if (idx >= Kin * Nout) return;
    int in = idx / Nout;
    int n  = idx % Nout;
    int inp = in;
    if (use_perm) {
        int h = in >> 6, j = in & 63;
        inp = perm[h] * HD + j;
    }
    Wt[(size_t)n * Kin + inp] = f2bf(W[idx]);
}

// ---------------------------------------------------------------------------
// WMMA GEMM: C[M,N] = A[M,K] @ B[K,N], A bf16 row-major, Bt = B^T bf16 [N,K].
// One wave per 16x64 output tile. mode: 0 = bf16 out row-major (scaled),
// 1 = bf16 out scattered into Vt[B][KV][64][P], 2 = f32 out + bias.
// ---------------------------------------------------------------------------
__global__ void __launch_bounds__(32)
gemm_wmma_kernel(const unsigned short* __restrict__ A,
                 const unsigned short* __restrict__ Bt,
                 void* __restrict__ Cout, const float* __restrict__ bias,
                 int M, int N, int K, float scale, int mode) {
    const int lane = threadIdx.x & 31;
    const int lr = lane & 15;
    const int lh = lane >> 4;
    const int m0 = blockIdx.x * 16;
    const int n0 = blockIdx.y * 64;

    v8f acc[4];
#pragma unroll
    for (int t = 0; t < 4; ++t) acc[t] = vzero8();

    const unsigned short* arow = A + (size_t)(m0 + lr) * K;
    for (int k0 = 0; k0 < K; k0 += 32) {
        v16bf af = load_a_frag(arow + k0, lh);
#pragma unroll
        for (int t = 0; t < 4; ++t) {
            const unsigned short* bcol = Bt + (size_t)(n0 + t * 16 + lr) * K + k0;
            v16bf bf = load_b_frag(bcol + lh * 16);
            acc[t] = wmma_bf16(af, bf, acc[t]);
        }
    }

    // D layout: lane&15 = N, m = r + 8*(lane>>4)
    if (mode == 0) {
        unsigned short* C = (unsigned short*)Cout;
#pragma unroll
        for (int t = 0; t < 4; ++t)
#pragma unroll
            for (int r = 0; r < 8; ++r) {
                int m = m0 + r + 8 * lh;
                int n = n0 + t * 16 + lr;
                C[(size_t)m * N + n] = f2bf(acc[t][r] * scale);
            }
    } else if (mode == 1) {
        // scatter to Vt: [B][KV][64][P] ; m = b*P + p, n = kv*64 + j
        unsigned short* C = (unsigned short*)Cout;
#pragma unroll
        for (int t = 0; t < 4; ++t)
#pragma unroll
            for (int r = 0; r < 8; ++r) {
                int m = m0 + r + 8 * lh;
                int n = n0 + t * 16 + lr;
                int b = m >> 11, p = m & 2047;
                int kv = n >> 6, j = n & 63;
                C[((size_t)((b * NKV + kv) * HD + j)) * P_SEQ + p] = f2bf(acc[t][r]);
            }
    } else {
        float* C = (float*)Cout;
#pragma unroll
        for (int t = 0; t < 4; ++t)
#pragma unroll
            for (int r = 0; r < 8; ++r) {
                int m = m0 + r + 8 * lh;
                int n = n0 + t * 16 + lr;
                C[(size_t)m * N + n] = acc[t][r] + bias[n];
            }
    }
}

// ---------------------------------------------------------------------------
// Flash attention: one wave per (b, h, 16-query-row tile). Streams keys in
// 32-key chunks with online softmax. K/V chunks are staged in LDS with
// double-buffered global_load_async_to_lds_b128 (ASYNCcnt), so the WMMA
// stream reads from LDS and the next chunk's HBM/L2 latency is hidden.
//   Q  : bf16 [B,P,768]  (pre-scaled by d^-0.5)
//   Kb : bf16 [B,P,256]
//   Vt : bf16 [B,KV,64,P]
//   Att: bf16 [B,P,768]  (head-major output, ready for final projection)
// ---------------------------------------------------------------------------
__global__ void __launch_bounds__(32)
flash_attn_kernel(const unsigned short* __restrict__ Q,
                  const unsigned short* __restrict__ Kb,
                  const unsigned short* __restrict__ Vt,
                  unsigned short* __restrict__ Att) {
    __shared__ __align__(32) unsigned short s_k[2][32 * 64]; // 2 x 4KB K chunk
    __shared__ __align__(32) unsigned short s_v[2][64 * 32]; // 2 x 4KB V chunk
    __shared__ float          s_s[16 * 32];   // f32 score tile
    __shared__ unsigned short s_p[16 * 32];   // bf16 softmax probs (A-layout src)
    __shared__ float          s_a[16];        // per-row rescale alpha
    __shared__ float          s_l[16];        // per-row 1/l

    const int lane = threadIdx.x & 31;
    const int lr = lane & 15;
    const int lh = lane >> 4;
    const int b  = blockIdx.z;
    const int h  = blockIdx.y;
    const int kv = h / GRP;
    const int m0 = blockIdx.x * 16;

    const unsigned short* qrow  = Q  + ((size_t)(b * P_SEQ + m0 + lr) * CDIM) + h * HD;
    const unsigned short* kbase = Kb + ((size_t)(b * P_SEQ) * KVDIM) + kv * HD;
    const unsigned short* vbase = Vt + ((size_t)(b * NKV + kv) * HD) * P_SEQ;

    // Q fragments for the two K=32 steps covering d=64
    v16bf qa0 = load_a_frag(qrow, lh);
    v16bf qa1 = load_a_frag(qrow + 32, lh);

    float m_i = -1e30f, l_i = 0.f;   // valid in lanes 0..15 (row = lane)
    v8f o[4];
#pragma unroll
    for (int t = 0; t < 4; ++t) o[t] = vzero8();

    // prologue: async-stage chunk 0 into buffer 0
    async_copy_kv_chunk(kbase, vbase, s_k[0], s_v[0], 0, lane);

    int buf = 0;
    for (int kb0 = 0; kb0 < P_SEQ; kb0 += 32) {
        const bool last = (kb0 + 32 >= P_SEQ);
        // kick off next chunk into the other buffer before waiting
        if (!last)
            async_copy_kv_chunk(kbase, vbase, s_k[buf ^ 1], s_v[buf ^ 1],
                                kb0 + 32, lane);
        // current buffer's 16 async ops are the oldest outstanding; async
        // loads complete in order, so outstanding<=16 (the just-issued next
        // chunk) means this buffer is ready. Last chunk: drain fully.
        if (last) asm volatile("s_wait_asynccnt 0x0" ::: "memory");
        else      asm volatile("s_wait_asynccnt 0x10" ::: "memory");

        const unsigned short* kc = s_k[buf];
        const unsigned short* vc = s_v[buf];

        // ---- scores: S = Q @ K^T (two 16-key tiles, K-dim = d = 64) ----
        v8f s0 = vzero8(), s1 = vzero8();
        s0 = wmma_bf16(qa0, load_b_frag(kc + (size_t)lr * 64 + lh * 16), s0);
        s0 = wmma_bf16(qa1, load_b_frag(kc + (size_t)lr * 64 + 32 + lh * 16), s0);
        s1 = wmma_bf16(qa0, load_b_frag(kc + (size_t)(16 + lr) * 64 + lh * 16), s1);
        s1 = wmma_bf16(qa1, load_b_frag(kc + (size_t)(16 + lr) * 64 + 32 + lh * 16), s1);

        // stage f32 scores in LDS: s_s[m][n], m = r + 8*lh, n = lr (+16)
#pragma unroll
        for (int r = 0; r < 8; ++r) {
            s_s[(r + 8 * lh) * 32 + lr]      = s0[r];
            s_s[(r + 8 * lh) * 32 + 16 + lr] = s1[r];
        }
        __syncthreads();

        // ---- online softmax: lanes 0..15 each own one query row ----
        if (lane < 16) {
            float mx = m_i;
#pragma unroll
            for (int j = 0; j < 32; ++j) mx = fmaxf(mx, s_s[lane * 32 + j]);
            float alpha = __expf(m_i - mx);
            float sum = 0.f;
#pragma unroll
            for (int j = 0; j < 32; ++j) {
                float p = __expf(s_s[lane * 32 + j] - mx);
                sum += p;
                s_p[lane * 32 + j] = f2bf(p);
            }
            l_i = l_i * alpha + sum;
            m_i = mx;
            s_a[lane] = alpha;
        }
        __syncthreads();

        // rescale accumulator rows by alpha
        float ar[8];
#pragma unroll
        for (int r = 0; r < 8; ++r) ar[r] = s_a[r + 8 * lh];
#pragma unroll
        for (int t = 0; t < 4; ++t)
#pragma unroll
            for (int r = 0; r < 8; ++r) o[t][r] *= ar[r];

        // ---- O += P @ V : A = probs (16x32), B = V chunk (32 x 16 per tile)
        v16bf pa = load_a_frag(&s_p[lr * 32], lh);
#pragma unroll
        for (int t = 0; t < 4; ++t)
            o[t] = wmma_bf16(pa, load_b_frag(vc + (t * 16 + lr) * 32 + lh * 16),
                             o[t]);
        // next iteration's s_s writes are ordered by its first __syncthreads,
        // so two barriers per chunk suffice.
        buf ^= 1;
    }

    if (lane < 16) s_l[lane] = 1.0f / l_i;
    __syncthreads();
    float linv[8];
#pragma unroll
    for (int r = 0; r < 8; ++r) linv[r] = s_l[r + 8 * lh];

    unsigned short* obase = Att + ((size_t)(b * P_SEQ + m0) * CDIM) + h * HD;
#pragma unroll
    for (int t = 0; t < 4; ++t)
#pragma unroll
        for (int r = 0; r < 8; ++r) {
            float v = o[t][r] * linv[r];
            obase[(size_t)(r + 8 * lh) * CDIM + t * 16 + lr] = f2bf(v);
        }
}

// ---------------------------------------------------------------------------
// Launch
// ---------------------------------------------------------------------------
extern "C" void kernel_launch(void* const* d_in, const int* in_sizes, int n_in,
                              void* d_out, int out_size, void* d_ws, size_t ws_size,
                              hipStream_t stream) {
    (void)in_sizes; (void)n_in; (void)out_size; (void)ws_size;
    const float* x    = (const float*)d_in[0];
    // d_in[1] = exp_num (unused)
    const int*   perm = (const int*)d_in[2];
    const float* Wq   = (const float*)d_in[3];
    const float* Wk   = (const float*)d_in[4];
    const float* Wv   = (const float*)d_in[5];
    const float* Wp   = (const float*)d_in[6];
    const float* bp   = (const float*)d_in[7];
    float* out = (float*)d_out;

    char* ws = (char*)d_ws;
    size_t off = 0;
    auto alloc_us = [&](size_t nelem) -> unsigned short* {
        off = (off + 255) & ~(size_t)255;
        unsigned short* p = (unsigned short*)(ws + off);
        off += nelem * sizeof(unsigned short);
        return p;
    };

    unsigned short* xb   = alloc_us((size_t)MTOT * CDIM);     // x bf16
    unsigned short* Wqt  = alloc_us((size_t)CDIM * CDIM);     // Wq^T (perm rows)
    unsigned short* Wkt  = alloc_us((size_t)KVDIM * CDIM);    // Wk^T (perm rows)
    unsigned short* Wvt  = alloc_us((size_t)KVDIM * CDIM);    // Wv^T (perm rows)
    unsigned short* Wpt  = alloc_us((size_t)CDIM * CDIM);     // Wproj^T
    unsigned short* Qb   = alloc_us((size_t)MTOT * CDIM);     // Q (scaled) bf16
    unsigned short* Kbuf = alloc_us((size_t)MTOT * KVDIM);    // K bf16
    unsigned short* VtB  = alloc_us((size_t)B_SZ * NKV * HD * P_SEQ); // V^T
    unsigned short* Att  = alloc_us((size_t)MTOT * CDIM);     // attention out bf16

    const int nx = MTOT * CDIM;
    conv_bf16_kernel<<<(nx + 255) / 256, 256, 0, stream>>>(x, xb, nx);

    prep_w_kernel<<<(CDIM * CDIM + 255) / 256, 256, 0, stream>>>(
        Wq, Wqt, perm, CDIM, CDIM, 1);
    prep_w_kernel<<<(CDIM * KVDIM + 255) / 256, 256, 0, stream>>>(
        Wk, Wkt, perm, CDIM, KVDIM, 1);
    prep_w_kernel<<<(CDIM * KVDIM + 255) / 256, 256, 0, stream>>>(
        Wv, Wvt, perm, CDIM, KVDIM, 1);
    prep_w_kernel<<<(CDIM * CDIM + 255) / 256, 256, 0, stream>>>(
        Wp, Wpt, perm, CDIM, CDIM, 0);

    const float qscale = 0.125f; // 1/sqrt(64)
    gemm_wmma_kernel<<<dim3(MTOT / 16, CDIM / 64), 32, 0, stream>>>(
        xb, Wqt, Qb, nullptr, MTOT, CDIM, CDIM, qscale, 0);
    gemm_wmma_kernel<<<dim3(MTOT / 16, KVDIM / 64), 32, 0, stream>>>(
        xb, Wkt, Kbuf, nullptr, MTOT, KVDIM, CDIM, 1.0f, 0);
    gemm_wmma_kernel<<<dim3(MTOT / 16, KVDIM / 64), 32, 0, stream>>>(
        xb, Wvt, VtB, nullptr, MTOT, KVDIM, CDIM, 1.0f, 1);

    flash_attn_kernel<<<dim3(P_SEQ / 16, NHEAD, B_SZ), 32, 0, stream>>>(
        Qb, Kbuf, VtB, Att);

    gemm_wmma_kernel<<<dim3(MTOT / 16, CDIM / 64), 32, 0, stream>>>(
        Att, Wpt, out, bp, MTOT, CDIM, CDIM, 1.0f, 2);
}
